// CAMHSA_37538014167550
// MI455X (gfx1250) — matrix operand order repaired and checked by reading, not compiled, not run on hardware
//
#include <hip/hip_runtime.h>
#include <hip/hip_bf16.h>
#include <cstdint>

// ---------------------------------------------------------------------------
// CAMHSA for gfx1250 (MI455X).
//  - All 5 GEMM stages: V_WMMA_F32_16X16X32_F16, f16 operands pre-staged in
//    WMMA-friendly layouts so every fragment is a contiguous b128 load.
//  - Convs: fp32 VALU, weights staged into LDS via global_load_async_to_lds.
// ---------------------------------------------------------------------------

typedef __attribute__((ext_vector_type(16))) _Float16 v16h;
typedef __attribute__((ext_vector_type(8)))  _Float16 v8h;
typedef __attribute__((ext_vector_type(8)))  float    v8f;

constexpr int Bc  = 8;
constexpr int Nc  = 512;
constexpr int Dc  = 512;
constexpr int Hc  = 8;
constexpr int DHc = 64;
constexpr int PCc = 60;
constexpr int RR  = 121;   // 2*PC+1

__device__ __forceinline__ v8f wmma_f16(v16h a, v16h b, v8f c) {
  // 8 args: (neg_a, A, neg_b, B, c_mod, C, reuse_a, reuse_b)
  return __builtin_amdgcn_wmma_f32_16x16x32_f16(false, a, false, b,
                                                (short)0, c, false, false);
}

// Assemble the 16-half A fragment from its two contiguous 8-half chunks.
__device__ __forceinline__ v16h afrag(const _Float16* ap) {
  v8h lo = *(const v8h*)(ap);        // K = base .. base+7
  v8h hi = *(const v8h*)(ap + 8);    // K = base+8 .. base+15
  return __builtin_shufflevector(lo, hi, 0, 1, 2, 3, 4, 5, 6, 7,
                                         8, 9, 10, 11, 12, 13, 14, 15);
}

// Async-stage `n` floats of weights into LDS (per CDNA5 ISA).
__device__ __forceinline__ void stage_lds_async(float* sw, const float* w, int n) {
  for (int i = threadIdx.x; i < n; i += 256) {
    unsigned ldsoff = (unsigned)(uintptr_t)(&sw[i]);
    const float* gp = w + i;
    asm volatile("global_load_async_to_lds_b32 %0, %1, off"
                 :: "v"(ldsoff), "v"(gp) : "memory");
  }
  asm volatile("s_wait_asynccnt 0" ::: "memory");
  __syncthreads();
}

// ---------------------------------------------------------------------------
// K0a: query -> f16 copy (row-major)
// ---------------------------------------------------------------------------
__global__ __launch_bounds__(256) void k_cvtX(const float* __restrict__ x,
                                              _Float16* __restrict__ x16) {
  const long i = (long)blockIdx.x * 256 + threadIdx.x;   // B*N*D exact
  x16[i] = (_Float16)x[i];
}

// K0b: four weights -> f16 TRANSPOSED: WT[n*512+k] = W[k*512+n]
__global__ __launch_bounds__(256) void k_cvtW(
    const float* __restrict__ w0, const float* __restrict__ w1,
    const float* __restrict__ w2, const float* __restrict__ w3,
    _Float16* __restrict__ t0, _Float16* __restrict__ t1,
    _Float16* __restrict__ t2, _Float16* __restrict__ t3) {
  const long idx = (long)blockIdx.x * 256 + threadIdx.x;  // 4*512*512 exact
  const int which = (int)(idx >> 18);
  const int rem   = (int)(idx & 262143);
  const int n = rem >> 9, k = rem & 511;
  const float* w = (which == 0) ? w0 : (which == 1) ? w1 : (which == 2) ? w2 : w3;
  _Float16*   t = (which == 0) ? t0 : (which == 1) ? t1 : (which == 2) ? t2 : t3;
  t[(long)n * Dc + k] = (_Float16)w[(long)k * Dc + n];
}

// ---------------------------------------------------------------------------
// K1: fused QKV projections, one wave per 16x32 tile (2 accumulators).
//   Q -> f32 (for pos-table) + f16 row-major
//   K -> f16 row-major          V -> f16 TRANSPOSED [b, dcol, m]
// ---------------------------------------------------------------------------
__global__ __launch_bounds__(256) void k_gemm_qkv(
    const _Float16* __restrict__ X16, const float* __restrict__ bq,
    const _Float16* __restrict__ WqT, const _Float16* __restrict__ WkT,
    const _Float16* __restrict__ WvT,
    const float* __restrict__ bk, const float* __restrict__ bv,
    float* __restrict__ Qf, _Float16* __restrict__ Q16,
    _Float16* __restrict__ K16, _Float16* __restrict__ VT16)
{
  const int wave = threadIdx.x >> 5;
  const int lane = threadIdx.x & 31;
  const long tile = (long)blockIdx.x * 8 + wave;   // 3 * 256 mtiles * 16 npairs
  const int which = (int)(tile / 4096);            // 0=Q 1=K 2=V
  const int rem   = (int)(tile % 4096);
  const int mtile = rem >> 4;
  const int npair = rem & 15;

  const _Float16* WT   = (which == 0) ? WqT : (which == 1) ? WkT : WvT;
  const float*    bias = (which == 0) ? bq  : (which == 1) ? bk  : bv;

  const int halfsel = lane >> 4;
  const int lid     = lane & 15;
  const int arow  = mtile * 16 + lid;
  const int bcol0 = npair * 32 + lid;
  const int bcol1 = bcol0 + 16;

  v8f acc0 = {}, acc1 = {};
  for (int k0 = 0; k0 < Dc; k0 += 32) {
    const _Float16* ap = X16 + (long)arow * Dc + k0 + halfsel * 16;
    __builtin_prefetch(ap + 64, 0, 0);
    const v16h a  = afrag(ap);
    const int  kB = k0 + halfsel * 16;
    const v16h b0 = *(const v16h*)(WT + (long)bcol0 * Dc + kB);
    const v16h b1 = *(const v16h*)(WT + (long)bcol1 * Dc + kB);
    acc0 = wmma_f16(a, b0, acc0);
    acc1 = wmma_f16(a, b1, acc1);
  }
  const float bb0 = bias[bcol0], bb1 = bias[bcol1];
  const int rowb = mtile * 16 + 8 * halfsel;

  if (which == 0) {
#pragma unroll
    for (int r = 0; r < 8; ++r) {
      const long o0 = (long)(rowb + r) * Dc + bcol0;
      const long o1 = (long)(rowb + r) * Dc + bcol1;
      const float v0 = acc0[r] + bb0, v1 = acc1[r] + bb1;
      Qf[o0] = v0;  Qf[o1] = v1;
      Q16[o0] = (_Float16)v0;  Q16[o1] = (_Float16)v1;
    }
  } else if (which == 1) {
#pragma unroll
    for (int r = 0; r < 8; ++r) {
      K16[(long)(rowb + r) * Dc + bcol0] = (_Float16)(acc0[r] + bb0);
      K16[(long)(rowb + r) * Dc + bcol1] = (_Float16)(acc1[r] + bb1);
    }
  } else {
    v8h p0, p1;
#pragma unroll
    for (int r = 0; r < 8; ++r) {
      p0[r] = (_Float16)(acc0[r] + bb0);
      p1[r] = (_Float16)(acc1[r] + bb1);
    }
    const int b_ = rowb >> 9, ml = rowb & 511;        // tile never crosses b
    *(v8h*)(VT16 + ((long)b_ * Dc + bcol0) * Nc + ml) = p0;
    *(v8h*)(VT16 + ((long)b_ * Dc + bcol1) * Nc + ml) = p1;
  }
}

// ---------------------------------------------------------------------------
// K2: relative-position dot table  qp[b,n,h,r] = (q[b,n,h,:]·pos_k[r,:]) / 8
// ---------------------------------------------------------------------------
__global__ __launch_bounds__(256) void k_qpos(
    const float* __restrict__ Q, const float* __restrict__ posk,
    float* __restrict__ QP)
{
  const long total = (long)Bc * Nc * Hc * RR;
  long idx = (long)blockIdx.x * 256 + threadIdx.x;
  if (idx >= total) return;
  const int r = (int)(idx % RR);
  long t = idx / RR;
  const int h = (int)(t % Hc); t /= Hc;
  const int n = (int)(t % Nc);
  const int b = (int)(t / Nc);
  const float* q = Q + ((long)(b * Nc + n)) * Dc + h * DHc;
  const float* p = posk + (long)r * DHc;
  float s = 0.f;
#pragma unroll 8
  for (int d = 0; d < DHc; ++d) s = fmaf(q[d], p[d], s);
  QP[idx] = s * 0.125f;            // fold 1/sqrt(DH)
}

// ---------------------------------------------------------------------------
// K3: scores tile = (q·kᵀ)/8 + qp[clip(m-n)]  ->  maps[b,n,m,h] (NHWC)
// ---------------------------------------------------------------------------
__global__ __launch_bounds__(256) void k_scores(
    const _Float16* __restrict__ Q16, const _Float16* __restrict__ K16,
    const float* __restrict__ QP, float* __restrict__ maps)
{
  const int wave = threadIdx.x >> 5;
  const int lane = threadIdx.x & 31;
  const long tile = (long)blockIdx.x * 8 + wave;      // 64 hb * 32 * 32 tiles
  const int hb    = (int)(tile >> 10);
  const int rem   = (int)(tile & 1023);
  const int ntile = rem >> 5;
  const int mtile = rem & 31;
  const int h = hb >> 3, b = hb & 7;

  const int halfsel = lane >> 4;
  const int lid     = lane & 15;
  const int nrow = ntile * 16 + lid;
  const int mcol = mtile * 16 + lid;

  v8f acc = {};
#pragma unroll
  for (int k0 = 0; k0 < DHc; k0 += 32) {
    const v16h a = afrag(Q16 + ((long)(b * Nc + nrow)) * Dc + h * DHc + k0 + halfsel * 16);
    // B = kᵀ: column m fixed -> contiguous along K in k16's row
    const v16h bf = *(const v16h*)(K16 + ((long)(b * Nc + mcol)) * Dc
                                   + h * DHc + k0 + halfsel * 16);
    acc = wmma_f16(a, bf, acc);
  }
#pragma unroll
  for (int r = 0; r < 8; ++r) {
    const int ng = ntile * 16 + r + 8 * halfsel;
    int rel = mcol - ng;
    rel = rel < -PCc ? -PCc : (rel > PCc ? PCc : rel);
    rel += PCc;
    maps[(((long)(b * Nc + ng)) * Nc + mcol) * Hc + h] =
        acc[r] * 0.125f + QP[(((long)(b * Nc + ng)) * Hc + h) * RR + rel];
  }
}

// ---------------------------------------------------------------------------
// K4: conv1 5x5, 8 -> 16 channels, SAME, + bias.  Async LDS weight staging.
// ---------------------------------------------------------------------------
__global__ __launch_bounds__(256) void k_conv1(
    const float* __restrict__ inp, const float* __restrict__ w,
    const float* __restrict__ bias, float* __restrict__ out)
{
  __shared__ float sw[5 * 5 * 8 * 16];
  stage_lds_async(sw, w, 5 * 5 * 8 * 16);

  const long pix = (long)blockIdx.x * 256 + threadIdx.x;   // B*N*N pixels
  const int m = (int)(pix % Nc);
  long t = pix / Nc;
  const int n = (int)(t % Nc);
  const int b = (int)(t / Nc);

  float acc[16];
#pragma unroll
  for (int c = 0; c < 16; ++c) acc[c] = bias[c];

  for (int dn = 0; dn < 5; ++dn) {
    const int nn = n + dn - 2; if (nn < 0 || nn >= Nc) continue;
    for (int dm = 0; dm < 5; ++dm) {
      const int mm = m + dm - 2; if (mm < 0 || mm >= Nc) continue;
      const float* px = inp + (((long)(b * Nc + nn)) * Nc + mm) * Hc;
      const float* wt = sw + (dn * 5 + dm) * 8 * 16;
#pragma unroll
      for (int ci = 0; ci < 8; ++ci) {
        const float xv = px[ci];
#pragma unroll
        for (int c = 0; c < 16; ++c) acc[c] = fmaf(xv, wt[ci * 16 + c], acc[c]);
      }
    }
  }
  float* o = out + (((long)(b * Nc + n)) * Nc + m) * 16;
#pragma unroll
  for (int c = 0; c < 16; ++c) o[c] = acc[c];
}

// ---------------------------------------------------------------------------
// K5: masked group-norm statistics (rows n < valid_len[b], all m, per (b,c))
// ---------------------------------------------------------------------------
__global__ void k_zero(float* p, int n) {
  int i = blockIdx.x * 256 + threadIdx.x;
  if (i < n) p[i] = 0.f;
}

__global__ __launch_bounds__(256) void k_stats(
    const float* __restrict__ y, const int* __restrict__ vlen,
    float* __restrict__ ssum, float* __restrict__ ssq)
{
  const int b = blockIdx.x / Nc, n = blockIdx.x % Nc;
  if (n >= vlen[b]) return;
  const int c     = threadIdx.x & 15;
  const int mlane = threadIdx.x >> 4;
  float s = 0.f, s2 = 0.f;
  const float* row = y + (((long)(b * Nc + n)) * Nc) * 16;
  for (int m = mlane; m < Nc; m += 16) {
    const float v = row[(long)m * 16 + c];
    s += v; s2 = fmaf(v, v, s2);
  }
  __shared__ float sh[256], sh2[256];
  sh[threadIdx.x] = s; sh2[threadIdx.x] = s2;
  __syncthreads();
  if (threadIdx.x < 16) {
    float a = 0.f, a2 = 0.f;
    for (int i = threadIdx.x; i < 256; i += 16) { a += sh[i]; a2 += sh2[i]; }
    atomicAdd(&ssum[b * 16 + threadIdx.x], a);
    atomicAdd(&ssq [b * 16 + threadIdx.x], a2);
  }
}

__global__ void k_finalize(const float* __restrict__ ssum,
                           const float* __restrict__ ssq,
                           const int* __restrict__ vlen,
                           float* __restrict__ mean, float* __restrict__ inv)
{
  const int i = blockIdx.x * 128 + threadIdx.x;
  if (i >= 128) return;
  const int b = i >> 4;
  const float cnt = (float)vlen[b] * (float)Nc;
  const float mu  = ssum[i] / cnt;
  const float var = ssq[i] / cnt - mu * mu;
  mean[i] = mu;
  inv[i]  = rsqrtf(var + 1e-7f);
}

// K6: y = relu(gamma*(y-mean)*inv + beta)
__global__ __launch_bounds__(256) void k_normrelu(
    float* __restrict__ y, const float* __restrict__ mean,
    const float* __restrict__ inv, const float* __restrict__ gamma,
    const float* __restrict__ beta)
{
  const long idx = (long)blockIdx.x * 256 + threadIdx.x;  // B*N*N*16 exact
  const int c = (int)(idx & 15);
  const int b = (int)((idx >> 4) / ((long)Nc * Nc));
  float v = y[idx];
  v = gamma[c] * (v - mean[b * 16 + c]) * inv[b * 16 + c] + beta[c];
  y[idx] = v > 0.f ? v : 0.f;
}

// ---------------------------------------------------------------------------
// K7: conv2 5x5, 16 -> 8 channels, SAME, + bias -> scores2[b,n,m,h]
// ---------------------------------------------------------------------------
__global__ __launch_bounds__(256) void k_conv2(
    const float* __restrict__ inp, const float* __restrict__ w,
    const float* __restrict__ bias, float* __restrict__ out)
{
  __shared__ float sw[5 * 5 * 16 * 8];
  stage_lds_async(sw, w, 5 * 5 * 16 * 8);

  const long pix = (long)blockIdx.x * 256 + threadIdx.x;
  const int m = (int)(pix % Nc);
  long t = pix / Nc;
  const int n = (int)(t % Nc);
  const int b = (int)(t / Nc);

  float acc[8];
#pragma unroll
  for (int c = 0; c < 8; ++c) acc[c] = bias[c];

  for (int dn = 0; dn < 5; ++dn) {
    const int nn = n + dn - 2; if (nn < 0 || nn >= Nc) continue;
    for (int dm = 0; dm < 5; ++dm) {
      const int mm = m + dm - 2; if (mm < 0 || mm >= Nc) continue;
      const float* px = inp + (((long)(b * Nc + nn)) * Nc + mm) * 16;
      const float* wt = sw + (dn * 5 + dm) * 16 * 8;
#pragma unroll
      for (int ci = 0; ci < 16; ++ci) {
        const float xv = px[ci];
#pragma unroll
        for (int c = 0; c < 8; ++c) acc[c] = fmaf(xv, wt[ci * 8 + c], acc[c]);
      }
    }
  }
  float* o = out + (((long)(b * Nc + n)) * Nc + m) * Hc;
#pragma unroll
  for (int c = 0; c < 8; ++c) o[c] = acc[c];
}

// ---------------------------------------------------------------------------
// K8a: column-masked softmax over m -> attn16[(h*B+b), n, m]  (f16 out)
// ---------------------------------------------------------------------------
__global__ __launch_bounds__(256) void k_softmax(
    const float* __restrict__ s2, const int* __restrict__ vlen,
    _Float16* __restrict__ attn16)
{
  const int h = blockIdx.x & 7;
  const long t = blockIdx.x >> 3;
  const int n = (int)(t % Nc), b = (int)(t / Nc);
  const int vl = vlen[b];
  const float* row = s2 + (((long)(b * Nc + n)) * Nc) * Hc + h;

  const int m0 = threadIdx.x, m1 = threadIdx.x + 256;
  const float x0 = (m0 < vl) ? row[(long)m0 * Hc] : -1e30f;
  const float x1 = (m1 < vl) ? row[(long)m1 * Hc] : -1e30f;

  __shared__ float sh[256];
  sh[threadIdx.x] = fmaxf(x0, x1);
  __syncthreads();
  for (int off = 128; off > 0; off >>= 1) {
    if (threadIdx.x < off) sh[threadIdx.x] = fmaxf(sh[threadIdx.x], sh[threadIdx.x + off]);
    __syncthreads();
  }
  const float mx = sh[0];
  __syncthreads();

  const float e0 = (m0 < vl) ? __expf(x0 - mx) : 0.f;
  const float e1 = (m1 < vl) ? __expf(x1 - mx) : 0.f;
  sh[threadIdx.x] = e0 + e1;
  __syncthreads();
  for (int off = 128; off > 0; off >>= 1) {
    if (threadIdx.x < off) sh[threadIdx.x] += sh[threadIdx.x + off];
    __syncthreads();
  }
  const float dn = 1.f / sh[0];
  _Float16* arow = attn16 + (((long)(h * Bc + b)) * Nc + n) * Nc;
  arow[m0] = (_Float16)(e0 * dn);
  arow[m1] = (_Float16)(e1 * dn);
}

// ---------------------------------------------------------------------------
// K8b: ctx16[b,n,h*64+d] = sum_m attn[hb,n,m] * v[b,m,h*64+d]   (WMMA)
// ---------------------------------------------------------------------------
__global__ __launch_bounds__(256) void k_av(
    const _Float16* __restrict__ attn16, const _Float16* __restrict__ VT16,
    _Float16* __restrict__ ctx16)
{
  const int wave = threadIdx.x >> 5;
  const int lane = threadIdx.x & 31;
  const long tile = (long)blockIdx.x * 8 + wave;   // 64 hb * 32 mtiles * 4 dtiles
  const int hb    = (int)(tile >> 7);
  const int rem   = (int)(tile & 127);
  const int mtile = rem >> 2;
  const int dtile = rem & 3;
  const int h = hb >> 3, b = hb & 7;

  const int halfsel = lane >> 4;
  const int lid     = lane & 15;
  const int nrow = mtile * 16 + lid;
  const int dcol = dtile * 16 + lid;
  const _Float16* abase = attn16 + ((long)(h * Bc + b)) * Nc * Nc;
  const _Float16* vbase = VT16 + ((long)b * Dc + h * DHc + dcol) * Nc;

  v8f acc = {};
  for (int k0 = 0; k0 < Nc; k0 += 32) {
    const v16h a  = afrag(abase + (long)nrow * Nc + k0 + halfsel * 16);
    const v16h bf = *(const v16h*)(vbase + k0 + halfsel * 16);  // contiguous K
    acc = wmma_f16(a, bf, acc);
  }
#pragma unroll
  for (int r = 0; r < 8; ++r) {
    const int ng = mtile * 16 + r + 8 * halfsel;
    ctx16[((long)(b * Nc + ng)) * Dc + h * DHc + dcol] = (_Float16)acc[r];
  }
}

// ---------------------------------------------------------------------------
// K9a: tmp = ctx @ Wh + bh + query  (WMMA, 16x32 tile/wave, residual fused)
// ---------------------------------------------------------------------------
__global__ __launch_bounds__(256) void k_gemm_out(
    const _Float16* __restrict__ X16, const _Float16* __restrict__ WhT,
    const float* __restrict__ bh, const float* __restrict__ resid,
    float* __restrict__ out)
{
  const int wave = threadIdx.x >> 5;
  const int lane = threadIdx.x & 31;
  const long tile = (long)blockIdx.x * 8 + wave;    // 256 mtiles * 16 npairs
  const int mtile = (int)(tile >> 4);
  const int npair = (int)(tile & 15);

  const int halfsel = lane >> 4;
  const int lid     = lane & 15;
  const int arow  = mtile * 16 + lid;
  const int bcol0 = npair * 32 + lid;
  const int bcol1 = bcol0 + 16;

  v8f acc0 = {}, acc1 = {};
  for (int k0 = 0; k0 < Dc; k0 += 32) {
    const _Float16* ap = X16 + (long)arow * Dc + k0 + halfsel * 16;
    __builtin_prefetch(ap + 64, 0, 0);
    const v16h a  = afrag(ap);
    const int  kB = k0 + halfsel * 16;
    const v16h b0 = *(const v16h*)(WhT + (long)bcol0 * Dc + kB);
    const v16h b1 = *(const v16h*)(WhT + (long)bcol1 * Dc + kB);
    acc0 = wmma_f16(a, b0, acc0);
    acc1 = wmma_f16(a, b1, acc1);
  }
  const float bb0 = bh[bcol0], bb1 = bh[bcol1];
  const int rowb = mtile * 16 + 8 * halfsel;
#pragma unroll
  for (int r = 0; r < 8; ++r) {
    const long o0 = (long)(rowb + r) * Dc + bcol0;
    const long o1 = (long)(rowb + r) * Dc + bcol1;
    out[o0] = acc0[r] + bb0 + resid[o0];
    out[o1] = acc1[r] + bb1 + resid[o1];
  }
}

// K9b: row LayerNorm over D=512 -> final output
__global__ __launch_bounds__(256) void k_layernorm(
    const float* __restrict__ x, const float* __restrict__ gamma,
    const float* __restrict__ beta, float* __restrict__ out)
{
  const long row = blockIdx.x;
  const float* xr = x + row * Dc;
  const float v0 = xr[threadIdx.x], v1 = xr[threadIdx.x + 256];

  __shared__ float sh[256];
  sh[threadIdx.x] = v0 + v1;
  __syncthreads();
  for (int off = 128; off > 0; off >>= 1) {
    if (threadIdx.x < off) sh[threadIdx.x] += sh[threadIdx.x + off];
    __syncthreads();
  }
  const float mean = sh[0] * (1.f / Dc);
  __syncthreads();
  const float d0 = v0 - mean, d1 = v1 - mean;
  sh[threadIdx.x] = d0 * d0 + d1 * d1;
  __syncthreads();
  for (int off = 128; off > 0; off >>= 1) {
    if (threadIdx.x < off) sh[threadIdx.x] += sh[threadIdx.x + off];
    __syncthreads();
  }
  const float inv = rsqrtf(sh[0] * (1.f / Dc) + 1e-7f);
  out[row * Dc + threadIdx.x]       = gamma[threadIdx.x] * d0 * inv + beta[threadIdx.x];
  out[row * Dc + threadIdx.x + 256] = gamma[threadIdx.x + 256] * d1 * inv + beta[threadIdx.x + 256];
}

// ---------------------------------------------------------------------------
extern "C" void kernel_launch(void* const* d_in, const int* in_sizes, int n_in,
                              void* d_out, int out_size, void* d_ws, size_t ws_size,
                              hipStream_t stream) {
  const float* query = (const float*)d_in[0];
  const int*   vlen  = (const int*)  d_in[1];
  const float* Wq    = (const float*)d_in[2];
  const float* bq    = (const float*)d_in[3];
  const float* Wk    = (const float*)d_in[4];
  const float* bk    = (const float*)d_in[5];
  const float* Wv    = (const float*)d_in[6];
  const float* bv    = (const float*)d_in[7];
  const float* Wh    = (const float*)d_in[8];
  const float* bh    = (const float*)d_in[9];
  const float* posk  = (const float*)d_in[10];
  const float* c1w   = (const float*)d_in[11];
  const float* c1b   = (const float*)d_in[12];
  const float* n1g   = (const float*)d_in[13];
  const float* n1b   = (const float*)d_in[14];
  const float* c2w   = (const float*)d_in[15];
  const float* c2b   = (const float*)d_in[16];
  const float* lng   = (const float*)d_in[17];
  const float* lnb   = (const float*)d_in[18];
  float* outp = (float*)d_out;

  // ---- workspace carve-up (bytes, 256B aligned chunks) ----
  char* p = (char*)d_ws;
  auto alloc = [&](size_t bytes) -> char* {
    char* r = p; p += (bytes + 255) & ~(size_t)255; return r;
  };
  const long NE_BND = (long)Bc * Nc * Dc;          //  2,097,152 elems
  const long NE_QP  = (long)Bc * Nc * Hc * RR;     //  3,964,928
  const long NE_MAP = (long)Bc * Nc * Nc * Hc;     // 16,777,216
  const long NE_Y1  = (long)Bc * Nc * Nc * 16;     // 33,554,432
  const long NE_W   = (long)Dc * Dc;               //    262,144

  float*    Qf    = (float*)   alloc(NE_BND * 4);
  _Float16* X16   = (_Float16*)alloc(NE_BND * 2);
  _Float16* Q16   = (_Float16*)alloc(NE_BND * 2);
  _Float16* K16   = (_Float16*)alloc(NE_BND * 2);
  _Float16* VT16  = (_Float16*)alloc(NE_BND * 2);
  _Float16* ctx16 = (_Float16*)alloc(NE_BND * 2);
  _Float16* WqT   = (_Float16*)alloc(NE_W * 2);
  _Float16* WkT   = (_Float16*)alloc(NE_W * 2);
  _Float16* WvT   = (_Float16*)alloc(NE_W * 2);
  _Float16* WhT   = (_Float16*)alloc(NE_W * 2);
  float*    qp    = (float*)   alloc(NE_QP * 4);
  float*    maps  = (float*)   alloc(NE_MAP * 4);   // reused as attn16 later
  float*    y1    = (float*)   alloc(NE_Y1 * 4);
  float*    s2    = (float*)   alloc(NE_MAP * 4);
  float*    tmp   = (float*)   alloc(NE_BND * 4);
  float*    ssum  = (float*)   alloc(128 * 4);
  float*    ssq   = (float*)   alloc(128 * 4);
  float*    mu    = (float*)   alloc(128 * 4);
  float*    inv   = (float*)   alloc(128 * 4);
  _Float16* attn16 = (_Float16*)maps;               // maps dead after conv1

  // 0) f16 staging of activations + transposed weights
  k_cvtX<<<(int)(NE_BND / 256), 256, 0, stream>>>(query, X16);
  k_cvtW<<<(int)(4 * NE_W / 256), 256, 0, stream>>>(Wq, Wk, Wv, Wh,
                                                    WqT, WkT, WvT, WhT);
  // 1) QKV projections: 3 * 4096 tiles / 8 waves
  k_gemm_qkv<<<1536, 256, 0, stream>>>(X16, bq, WqT, WkT, WvT, bk, bv,
                                       Qf, Q16, K16, VT16);
  // 2) relative-position table
  k_qpos<<<(int)((NE_QP + 255) / 256), 256, 0, stream>>>(Qf, posk, qp);
  // 3) scores -> maps (NHWC); 65536 tiles / 8 waves
  k_scores<<<8192, 256, 0, stream>>>(Q16, K16, qp, maps);
  // 4) conv1 (+bias)
  k_conv1<<<8192, 256, 0, stream>>>(maps, c1w, c1b, y1);
  // 5) masked norm stats
  k_zero<<<1, 256, 0, stream>>>(ssum, 256);         // zeros ssum+ssq
  k_stats<<<Bc * Nc, 256, 0, stream>>>(y1, vlen, ssum, ssq);
  k_finalize<<<1, 128, 0, stream>>>(ssum, ssq, vlen, mu, inv);
  // 6) normalize + relu
  k_normrelu<<<131072, 256, 0, stream>>>(y1, mu, inv, n1g, n1b);
  // 7) conv2 (+bias)
  k_conv2<<<8192, 256, 0, stream>>>(y1, c2w, c2b, s2);
  // 8) masked softmax (f16 out), then attn @ V (WMMA)
  k_softmax<<<Bc * Nc * Hc, 256, 0, stream>>>(s2, vlen, attn16);
  k_av<<<1024, 256, 0, stream>>>(attn16, VT16, ctx16);
  // 9) output projection + residual, then LayerNorm
  k_gemm_out<<<512, 256, 0, stream>>>(ctx16, WhT, bh, query, tmp);
  k_layernorm<<<Bc * Nc, 256, 0, stream>>>(tmp, lng, lnb, outp);
}
